// FourierTransform_55860344651996
// MI455X (gfx1250) — matrix operand order repaired and checked by reading, not compiled, not run on hardware
//
#include <hip/hip_runtime.h>
#include <math.h>

typedef __attribute__((ext_vector_type(16))) __bf16          v16bf;
typedef __attribute__((ext_vector_type(8)))  float           v8f;

#define CCH 128
#define HW  65536            // 256*256
#define PI2_OVER_N 0.04908738521234052f  // 2*pi/128
#define ROWW 68              // LDS row stride in dwords (64 data + 4 pad)
#define PLANE (16 * ROWW)    // dwords per plane
#define COL_TILES 4          // 4 tiles of 16 cols -> 64 cols per block

struct U32x8 { uint4 a, b; };   // 32B carrier for bit_cast to v16bf

// ---------- bf16 split helpers ----------
__device__ __forceinline__ unsigned int bf16_rne_u(float f) {
    unsigned int u = __float_as_uint(f);
    u += 0x7fffu + ((u >> 16) & 1u);
    return u >> 16;
}
__device__ __forceinline__ float bf16_up(unsigned int h) {
    return __uint_as_float(h << 16);
}

// ---------- precompute 1: T1 = A * Re(D*S*F), T2 = P * Im(D*S*F) ----------
__global__ void build_T_kernel(const float* __restrict__ amp_w,
                               const float* __restrict__ phase_w,
                               const float* __restrict__ freq_mask,
                               float* __restrict__ T1,
                               float* __restrict__ T2) {
    const int o = blockIdx.x;   // output channel of the 1x1 convs
    const int n = threadIdx.x;  // original input channel
    float s1 = 0.f, s2 = 0.f;
    for (int s = 0; s < CCH; ++s) {
        const int k  = (s + 64) & 127;
        const float th = (float)((k * n) & 127) * PI2_OVER_N;  // exact mod-128 reduction
        float sn, cs;
        sincosf(th, &sn, &cs);
        const float d = freq_mask[s];
        s1 += amp_w[o * CCH + s]   * (d * cs);   // Re(e^{-i th}) =  cos
        s2 -= phase_w[o * CCH + s] * (d * sn);   // Im(e^{-i th}) = -sin
    }
    T1[o * CCH + n] = s1;
    T2[o * CCH + n] = s2;
}

// ---------- precompute 2: M = (1/128)(C*Pi*T1 - S*Pi*T2) as bf16 hi/lo planes ----------
__global__ void build_M_kernel(const float* __restrict__ T1,
                               const float* __restrict__ T2,
                               const float* __restrict__ amp_b,
                               const float* __restrict__ phase_b,
                               unsigned short* __restrict__ Mhi,
                               unsigned short* __restrict__ Mlo,
                               float* __restrict__ bias) {
    const int n = blockIdx.x;   // ifft output index (row of M)
    const int m = threadIdx.x;  // original input channel (col of M)
    float acc = 0.f;
    for (int k = 0; k < CCH; ++k) {
        const float th = (float)((k * n) & 127) * PI2_OVER_N;
        float sn, cs;
        sincosf(th, &sn, &cs);
        const int kp = (k + 64) & 127;   // ifftshift row perm
        acc += cs * T1[kp * CCH + m] - sn * T2[kp * CCH + m];
    }
    acc *= (1.0f / 128.0f);
    const unsigned int hi = bf16_rne_u(acc);
    const unsigned int lo = bf16_rne_u(acc - bf16_up(hi));
    Mhi[n * CCH + m] = (unsigned short)hi;
    Mlo[n * CCH + m] = (unsigned short)lo;

    if (m == 0) {
        float bacc = 0.f;
        for (int k = 0; k < CCH; ++k) {
            const float th = (float)((k * n) & 127) * PI2_OVER_N;
            float sn, cs;
            sincosf(th, &sn, &cs);
            const int kp = (k + 64) & 127;
            bacc += cs * amp_b[kp] - sn * phase_b[kp];
        }
        bias[n] = bacc * (1.0f / 128.0f);
    }
}

// ---------- main: out[b,:,hw] = relu(M @ x[b,:,hw] + bias) via bf16x3 WMMA ----------
__global__ void __launch_bounds__(256)
fourier_fused_gemm(const float* __restrict__ x,
                   const unsigned short* __restrict__ Mhi,
                   const unsigned short* __restrict__ Mlo,
                   const float* __restrict__ bias,
                   float* __restrict__ out) {
    // two bf16 planes, layout [n][c] with c contiguous (packed as dwords = c-pairs)
    __shared__ unsigned int lds[2 * PLANE];

    const int tid  = threadIdx.x;
    const int wave = tid >> 5;        // 0..7 -> output rows 16*wave..16*wave+15
    const int lane = tid & 31;
    const int ln   = lane & 15;       // N index / A row within tile
    const int lh   = lane >> 4;       // lane half

    const int col0 = blockIdx.x * (16 * COL_TILES);
    const int b    = col0 >> 16;           // col0 / HW
    const int hw0  = col0 & (HW - 1);
    const float* xbase = x   + (size_t)b * CCH * HW + hw0;
    float*       obase = out + (size_t)b * CCH * HW + hw0;

    // --- A fragments: rows 16*wave+ln of M, all K, resident in VGPRs, no unpack ---
    // frag dwords 0..3 <- 16B at K=32kb+8h ; dwords 4..7 <- 16B at K=32kb+16+8h
    v16bf a_hi[4], a_lo[4];
    {
        const int row = 16 * wave + ln;
#pragma unroll
        for (int kb = 0; kb < 4; ++kb) {
            const int k0 = 32 * kb + 8 * lh;
            U32x8 th = { *(const uint4*)(Mhi + row * CCH + k0),
                         *(const uint4*)(Mhi + row * CCH + k0 + 16) };
            U32x8 tl = { *(const uint4*)(Mlo + row * CCH + k0),
                         *(const uint4*)(Mlo + row * CCH + k0 + 16) };
            a_hi[kb] = __builtin_bit_cast(v16bf, th);
            a_lo[kb] = __builtin_bit_cast(v16bf, tl);
        }
    }

    // per-lane output-row biases (D element r -> row 16*wave + 8*lh + r)
    float bv[8];
#pragma unroll
    for (int r = 0; r < 8; ++r) bv[r] = bias[16 * wave + 8 * lh + r];

    for (int ct = 0; ct < COL_TILES; ++ct) {
        const int ctoff = ct * 16;

        // cooperative load + hi/lo split: 1024 c-pairs, packed dwords land final
#pragma unroll
        for (int i = 0; i < 4; ++i) {
            const int p     = tid + 256 * i;
            const int n     = p & 15;
            const int cpair = p >> 4;            // 0..63
            const float x0 = xbase[(size_t)(2 * cpair)     * HW + ctoff + n];
            const float x1 = xbase[(size_t)(2 * cpair + 1) * HW + ctoff + n];
            const unsigned int h0 = bf16_rne_u(x0);
            const unsigned int h1 = bf16_rne_u(x1);
            const unsigned int l0 = bf16_rne_u(x0 - bf16_up(h0));
            const unsigned int l1 = bf16_rne_u(x1 - bf16_up(h1));
            lds[n * ROWW + cpair]         = (h1 << 16) | h0;   // hi plane
            lds[PLANE + n * ROWW + cpair] = (l1 << 16) | l0;   // lo plane
        }
        __syncthreads();

        v8f acc = {0.f, 0.f, 0.f, 0.f, 0.f, 0.f, 0.f, 0.f};
#pragma unroll
        for (int kb = 0; kb < 4; ++kb) {
            // B fragment dwords 0..3 <- K = 32kb+8lh.., dwords 4..7 <- K = 32kb+16+8lh.. (matches A)
            const unsigned int* ph = lds + ln * ROWW + 16 * kb + 4 * lh;
            U32x8 th = { *(const uint4*)ph, *(const uint4*)(ph + 8) };
            U32x8 tl = { *(const uint4*)(ph + PLANE), *(const uint4*)(ph + PLANE + 8) };
            const v16bf bh = __builtin_bit_cast(v16bf, th);
            const v16bf bl = __builtin_bit_cast(v16bf, tl);
            // bf16x3: hi*hi + hi*lo + lo*hi, f32 accumulate
            acc = __builtin_amdgcn_wmma_f32_16x16x32_bf16(false, a_hi[kb], false, bh,
                                                          (short)0, acc, false, false);
            acc = __builtin_amdgcn_wmma_f32_16x16x32_bf16(false, a_hi[kb], false, bl,
                                                          (short)0, acc, false, false);
            acc = __builtin_amdgcn_wmma_f32_16x16x32_bf16(false, a_lo[kb], false, bh,
                                                          (short)0, acc, false, false);
        }

        // bias + relu + store (D element r -> row 16*wave + 8*lh + r, col ctoff+ln)
#pragma unroll
        for (int r = 0; r < 8; ++r) {
            const int m = 16 * wave + 8 * lh + r;
            float v = acc[r] + bv[r];
            v = v > 0.f ? v : 0.f;
            obase[(size_t)m * HW + ctoff + ln] = v;
        }
        __syncthreads();
    }
}

extern "C" void kernel_launch(void* const* d_in, const int* in_sizes, int n_in,
                              void* d_out, int out_size, void* d_ws, size_t ws_size,
                              hipStream_t stream) {
    const float* x         = (const float*)d_in[0];
    const float* freq_mask = (const float*)d_in[1];
    const float* amp_w     = (const float*)d_in[2];
    const float* amp_b     = (const float*)d_in[3];
    const float* phase_w   = (const float*)d_in[4];
    const float* phase_b   = (const float*)d_in[5];
    float* out = (float*)d_out;

    char* ws = (char*)d_ws;
    unsigned short* Mhi = (unsigned short*)ws;            // 32 KB bf16 hi plane of M
    unsigned short* Mlo = (unsigned short*)(ws + 32768);  // 32 KB bf16 lo plane of M
    float* bias = (float*)(ws + 65536);                   // 512 B
    float* T1   = (float*)(ws + 66048);                   // 64 KB
    float* T2   = (float*)(ws + 66048 + 65536);           // 64 KB

    build_T_kernel<<<dim3(CCH), dim3(CCH), 0, stream>>>(amp_w, phase_w, freq_mask, T1, T2);
    build_M_kernel<<<dim3(CCH), dim3(CCH), 0, stream>>>(T1, T2, amp_b, phase_b, Mhi, Mlo, bias);

    const int total_cols = 4 * HW;                        // B * H * W
    const int blocks = total_cols / (16 * COL_TILES);     // 4096
    fourier_fused_gemm<<<dim3(blocks), dim3(256), 0, stream>>>(x, Mhi, Mlo, bias, out);
}
